// SE3Transformer_62955630625388
// MI455X (gfx1250) — matrix operand order repaired and checked by reading, not compiled, not run on hardware
//
#include <hip/hip_runtime.h>
#include <math.h>
#include <stdint.h>

#define N_NODESC 4096
#define N_EDGESC 49152
#define N_GRAPHSC 64
#define ATOM_FC 32
#define NUM_CHC 16
#define C_MIDC 8
#define EPSILONC 1e-6f
#define BSTRIDE 1092
#define QTOTAL 7056

typedef __attribute__((ext_vector_type(16))) _Float16 v16h;
typedef __attribute__((ext_vector_type(8)))  float    v8f;
typedef __attribute__((ext_vector_type(4)))  unsigned int v4u;
typedef __attribute__((ext_vector_type(8)))  unsigned int v8u;

__host__ __device__ __forceinline__ int ndimd(int d){ return 2*d+1; }
__host__ __device__ __forceinline__ int iabs_(int a){ return a<0?-a:a; }
__host__ __device__ __forceinline__ int njd(int a,int b){ int m=a<b?a:b; return 2*m+1; }

__host__ __device__ inline int q_off(int DI,int DO,int J){
  int t=0;
  for(int di=0;di<4;di++) for(int dd=0;dd<4;dd++){
    int jmin=iabs_(di-dd), jmax=di+dd;
    for(int j=jmin;j<=jmax;j++){
      if(di==DI&&dd==DO&&j==J) return t;
      t += ndimd(dd)*ndimd(di)*(2*j+1);
    }
  }
  return t;
}
__host__ __device__ inline int b_off(int DI,int DO){
  int t=0;
  for(int di=0;di<4;di++) for(int dd=0;dd<4;dd++){
    if(di==DI&&dd==DO) return t;
    t += ndimd(dd)*ndimd(di)*njd(di,dd);
  }
  return t;
}

// ------------------------- Q table (Clebsch-Gordan) -------------------------
struct cplx{ double re, im; };
__device__ __forceinline__ cplx cmul(cplx a, cplx b){ cplx r; r.re=a.re*b.re-a.im*b.im; r.im=a.re*b.im+a.im*b.re; return r; }
__device__ double dfact(int n){ double r=1.0; for(int i=2;i<=n;i++) r*=(double)i; return r; }

__device__ double cgc(int j1,int m1,int j2,int m2,int j3,int m3){
  if(m1+m2!=m3) return 0.0;
  if(j3 < iabs_(j1-j2) || j3 > j1+j2) return 0.0;
  double pre = sqrt((2.0*j3+1.0)*dfact(j1+j2-j3)*dfact(j1-j2+j3)*dfact(-j1+j2+j3)/dfact(j1+j2+j3+1));
  pre *= sqrt(dfact(j3+m3)*dfact(j3-m3)*dfact(j1-m1)*dfact(j1+m1)*dfact(j2-m2)*dfact(j2+m2));
  int k0 = 0; if(j2-j3-m1>k0) k0=j2-j3-m1; if(j1-j3+m2>k0) k0=j1-j3+m2;
  int k1 = j1+j2-j3; if(j1-m1<k1) k1=j1-m1; if(j2+m2<k1) k1=j2+m2;
  double s=0.0;
  for(int k=k0;k<=k1;k++){
    double den = dfact(k)*dfact(j1+j2-j3-k)*dfact(j1-m1-k)*dfact(j2+m2-k)*dfact(j3-j2+m1+k)*dfact(j3-j1-m2+k);
    s += ((k&1)? -1.0:1.0)/den;
  }
  return pre*s;
}
__device__ cplx u_entry(int l,int r,int c){
  const double s2i = 0.70710678118654752440;
  int m = r - l;
  cplx z; z.re=0.0; z.im=0.0;
  if(m==0){ if(c==l) z.re=1.0; }
  else if(m>0){
    if(c==l+m) z.re = ((m&1)? -1.0:1.0)*s2i;
    else if(c==l-m) z.re = s2i;
  } else {
    int am=-m;
    if(c==l+m) z.im = s2i;
    else if(c==l-m) z.im = -(((am&1)?-1.0:1.0))*s2i;
  }
  return z;
}
__device__ cplx q_elem(int di,int J,int dd,int O,int i,int j){
  cplx acc; acc.re=0; acc.im=0;
  for(int M=0;M<=2*dd;M++){
    cplx uo = u_entry(dd,O,M); uo.im = -uo.im;  // conj
    if(uo.re==0.0 && uo.im==0.0) continue;
    for(int m=0;m<=2*di;m++){
      int n = J + (M-dd) - (m-di);
      if(n<0||n>2*J) continue;
      cplx ui = u_entry(di,i,m);
      if(ui.re==0.0&&ui.im==0.0) continue;
      cplx uj = u_entry(J,j,n);
      if(uj.re==0.0&&uj.im==0.0) continue;
      double c = cgc(di, m-di, J, n-J, dd, M-dd);
      if(c==0.0) continue;
      cplx t = cmul(cmul(uo,ui),uj);
      acc.re += t.re*c; acc.im += t.im*c;
    }
  }
  return acc;
}
__global__ void k_init_q(float* qtab){
  int tid = threadIdx.x;
  int cnt=0;
  for(int di=0;di<4;di++) for(int dd=0;dd<4;dd++){
    int jmin=iabs_(di-dd), jmax=di+dd;
    for(int J=jmin;J<=jmax;J++){
      if(cnt==tid){
        int no=ndimd(dd), ni=ndimd(di), nm=2*J+1;
        double mre=0, mim=0;
        for(int O=0;O<no;O++)for(int i=0;i<ni;i++)for(int j=0;j<nm;j++){
          cplx q = q_elem(di,J,dd,O,i,j);
          if(fabs(q.re)>mre) mre=fabs(q.re);
          if(fabs(q.im)>mim) mim=fabs(q.im);
        }
        int flip = (mim>mre)?1:0;
        int base = q_off(di,dd,J);
        for(int O=0;O<no;O++)for(int i=0;i<ni;i++)for(int j=0;j<nm;j++){
          cplx q = q_elem(di,J,dd,O,i,j);
          qtab[base + (O*ni+i)*nm + j] = (float)(flip? q.im : q.re);
        }
        return;
      }
      cnt++;
    }
  }
}

// ------------------------- basis + feat -------------------------
__global__ void k_basis(const float* __restrict__ pos, const int* __restrict__ src,
                        const int* __restrict__ dst, const float* __restrict__ edge_feat,
                        const float* __restrict__ qtab,
                        float* __restrict__ basis, float* __restrict__ feat){
  int e = blockIdx.x*blockDim.x + threadIdx.x;
  if(e >= N_EDGESC) return;
  int s = src[e], d = dst[e];
  float rx = pos[d*3+0]-pos[s*3+0];
  float ry = pos[d*3+1]-pos[s*3+1];
  float rz = pos[d*3+2]-pos[s*3+2];
  float r = sqrtf(rx*rx+ry*ry+rz*rz + 1e-12f);
  feat[e*5+0]=edge_feat[e*4+0]; feat[e*5+1]=edge_feat[e*4+1];
  feat[e*5+2]=edge_feat[e*4+2]; feat[e*5+3]=edge_feat[e*4+3];
  feat[e*5+4]=r;
  float x=rx/r, y=ry/r, z=rz/r;
  float st = sqrtf(fmaxf(x*x+y*y, 1e-12f));
  float cp = x/st, sp = y/st;
  float cosm[7], sinm[7];
  cosm[0]=1.f; sinm[0]=0.f; cosm[1]=cp; sinm[1]=sp;
  for(int k=2;k<7;k++){ cosm[k]=2.f*cp*cosm[k-1]-cosm[k-2]; sinm[k]=2.f*cp*sinm[k-1]-sinm[k-2]; }
  float P[7][7];
  P[0][0]=1.f;
  for(int m=1;m<7;m++) P[m][m] = -(2.f*m-1.f)*st*P[m-1][m-1];
  for(int m=0;m<6;m++) P[m+1][m] = (2.f*m+1.f)*z*P[m][m];
  for(int m=0;m<7;m++) for(int l=m+2;l<7;l++)
    P[l][m] = ((2.f*l-1.f)*z*P[l-1][m] - (float)(l+m-1)*P[l-2][m])/(float)(l-m);
  float Y[49];
  const double PI4 = 12.566370614359172954;
  for(int l=0;l<7;l++){
    for(int mm=-l;mm<=l;mm++){
      int mu = mm<0?-mm:mm;
      double fr = 1.0; for(int t=l-mu+1;t<=l+mu;t++) fr *= (double)t;   // (l+mu)!/(l-mu)!
      float Nn = (float)sqrt((2.0*l+1.0)/PI4/fr);
      float val;
      if(mm==0)      val = Nn*P[l][0];
      else if(mm>0)  val = 1.41421356237f*((mu&1)?-1.f:1.f)*Nn*P[l][mu]*cosm[mu];
      else           val = 1.41421356237f*((mu&1)?-1.f:1.f)*Nn*P[l][mu]*sinm[mu];
      Y[l*l + l + mm] = val;
    }
  }
  float* be = basis + (size_t)e*BSTRIDE;
  for(int di=0;di<4;di++) for(int dd=0;dd<4;dd++){
    int jmin=iabs_(di-dd), jmax=di+dd;
    int ni=ndimd(di), no=ndimd(dd), nJ=jmax-jmin+1;
    int bo = b_off(di,dd);
    for(int J=jmin;J<=jmax;J++){
      int qb = q_off(di,dd,J);
      int nm = 2*J+1;
      const float* Yl = Y + J*J;
      for(int o=0;o<no;o++) for(int i=0;i<ni;i++){
        float sacc=0.f;
        for(int m=0;m<nm;m++) sacc += qtab[qb+(o*ni+i)*nm+m]*Yl[m];
        be[bo + (o*ni+i)*nJ + (J-jmin)] = sacc;
      }
    }
  }
}

// ------------------------- radial MLP: WMMA + TDM -------------------------
// A-fragment K mapping (16-bit A 16x32, ISA 7.12.2)
__device__ __forceinline__ int kmapA(int t,int half){
  int v=t>>1, s=t&1;
  return (v<4)? (2*v+s+8*half) : (2*(v-4)+s+16+8*half);
}
// B-fragment K mapping: same as A (t<8: K=t+8*half, t>=8: K=t+8+8*half)
__device__ __forceinline__ int kmapB(int t,int half){ return (t<8)? (t + 8*half) : (t + 8 + 8*half); }

// Issue a TDM 2D tile load: 32 rows x 16 cols of f32 from w3 (row stride = Nout)
// into LDS at ldsAddr. D# per CDNA5 ISA sec 8.3/8.4 (count=1, data_size=4B, type=2).
__device__ __forceinline__ void tdm_load_tile(const float* gptr, int Nout, unsigned ldsAddr){
  unsigned long long ga = (unsigned long long)(uintptr_t)gptr;
  v4u g0;
  g0[0] = 1u;                                   // count=1 (valid descriptor)
  g0[1] = ldsAddr;                              // lds_addr [63:32]
  g0[2] = (unsigned)ga;                         // global_addr [95:64]
  g0[3] = ((unsigned)(ga>>32) & 0x01FFFFFFu) | (2u<<30);  // addr hi + type=2
  v8u g1;
  g1[0] = (2u<<16);                             // wg_mask=0, data_size=4B
  g1[1] = ((unsigned)Nout & 0xFFFFu) << 16;     // tensor_dim0[15:0]
  g1[2] = (((unsigned)Nout >> 16) & 0xFFFFu) | (32u<<16); // dim0 hi | tensor_dim1=32
  g1[3] = (16u<<16);                            // tensor_dim1 hi=0 | tile_dim0=16
  g1[4] = 32u;                                  // tile_dim1=32, tile_dim2=0
  g1[5] = (unsigned)Nout;                       // tensor_dim0_stride lo
  g1[6] = 0u;                                   // stride hi | tensor_dim1_stride lo
  g1[7] = 0u;
  v4u g2; g2[0]=0u; g2[1]=0u; g2[2]=0u; g2[3]=0u;
  v4u g3; g3[0]=0u; g3[1]=0u; g3[2]=0u; g3[3]=0u;
  asm volatile("tensor_load_to_lds %0, %1, %2, %3"
               :: "s"(g0), "s"(g1), "s"(g2), "s"(g3)
               : "memory");
}

__global__ void __launch_bounds__(256)
k_radial(const float* __restrict__ feat,
         const float* __restrict__ w1,const float* __restrict__ b1,
         const float* __restrict__ w2,const float* __restrict__ b2,
         const float* __restrict__ w3,const float* __restrict__ b3,
         int Nout, float* __restrict__ R){
  __shared__ _Float16 h1S[128*32];
  __shared__ _Float16 h2S[128*32];
  __shared__ _Float16 w2S[32*32];
  __shared__ float    w3S[8][2][512];   // per-wave double-buffered 32x16 f32 tile
  int tid = threadIdx.x;
  int eBase = blockIdx.x*128;
  // h1 = relu(feat @ w1 + b1)  (tiny K=5, VALU)
  for(int idx=tid; idx<128*32; idx+=256){
    int el = idx>>5, kk = idx&31;
    float acc = b1[kk];
    const float* f = feat + (size_t)(eBase+el)*5;
    #pragma unroll
    for(int i=0;i<5;i++) acc += f[i]*w1[i*32+kk];
    h1S[idx] = (_Float16)fmaxf(acc,0.f);
  }
  for(int idx=tid; idx<32*32; idx+=256) w2S[idx] = (_Float16)w2[idx];
  __syncthreads();
  int wv = tid>>5, lane = tid&31;
  int half = lane>>4, l16 = lane&15;
  int eloc = wv*16;
  // h2 = relu(h1 @ w2 + b2) via two 16x16x32 WMMAs
  v16h a1;
  #pragma unroll
  for(int t=0;t<16;t++) a1[t] = h1S[(eloc+l16)*32 + kmapA(t,half)];
  #pragma unroll
  for(int nt=0;nt<2;nt++){
    v16h b;
    #pragma unroll
    for(int t=0;t<16;t++) b[t] = w2S[kmapB(t,half)*32 + nt*16 + l16];
    v8f c = {};
    c = __builtin_amdgcn_wmma_f32_16x16x32_f16(false,a1,false,b,(short)0,c,false,false);
    #pragma unroll
    for(int v=0;v<8;v++){
      int row = v + 8*half, col = nt*16 + l16;
      h2S[(eloc+row)*32+col] = (_Float16)fmaxf(c[v] + b2[col], 0.f);
    }
  }
  __syncthreads();
  // R = h2 @ w3 + b3 via WMMA over N tiles; w3 tiles staged via async TDM
  v16h a2;
  #pragma unroll
  for(int t=0;t<16;t++) a2[t] = h2S[(eloc+l16)*32 + kmapA(t,half)];
  int ntiles = Nout>>4;
  unsigned lds0 = (unsigned)(uintptr_t)(&w3S[wv][0][0]);
  unsigned lds1 = (unsigned)(uintptr_t)(&w3S[wv][1][0]);
  tdm_load_tile(w3, Nout, lds0);
  for(int nt=0; nt<ntiles; nt++){
    int n0 = nt*16;
    if(nt+1 < ntiles){
      tdm_load_tile(w3 + (nt+1)*16, Nout, ((nt+1)&1)? lds1 : lds0);
      __builtin_amdgcn_s_wait_tensorcnt(1);   // current tile landed
    } else {
      __builtin_amdgcn_s_wait_tensorcnt(0);
    }
    const float* tile = &w3S[wv][nt&1][0];
    v16h b;
    #pragma unroll
    for(int t=0;t<16;t++) b[t] = (_Float16)tile[kmapB(t,half)*16 + l16];
    v8f c = {};
    c = __builtin_amdgcn_wmma_f32_16x16x32_f16(false,a2,false,b,(short)0,c,false,false);
    #pragma unroll
    for(int v=0;v<8;v++){
      int row = v + 8*half;
      R[(size_t)(eBase+eloc+row)*Nout + n0 + l16] = c[v] + b3[n0 + l16];
    }
  }
}

// ------------------------- pair message (basis x h_src x R) -------------------------
#define PM_WAVES 8
__global__ void __launch_bounds__(256)
k_pairmsg(const float* __restrict__ basis, int boff,
          const float* __restrict__ hsrc, const int* __restrict__ src,
          const float* __restrict__ R, int Nout,
          float* __restrict__ out, int acc,
          int di,int dd,int ci,int co){
  __shared__ float hsS[PM_WAVES][112];
  __shared__ float bS [PM_WAVES][343];
  __shared__ float tS [PM_WAVES][784];
  int wv = threadIdx.x>>5, lane = threadIdx.x&31;
  int e = blockIdx.x*PM_WAVES + wv;
  int ni = 2*di+1, no = 2*dd+1;
  int mn = di<dd?di:dd; int nJ = 2*mn+1;
  int s = src[e];
  int nhs = ci*ni;
  for(int idx=lane; idx<nhs; idx+=32) hsS[wv][idx] = hsrc[(size_t)s*nhs + idx];
  int nb = no*ni*nJ;
  const float* be = basis + (size_t)e*BSTRIDE + boff;
  for(int idx=lane; idx<nb; idx+=32) bS[wv][idx] = be[idx];
  __syncthreads();
  int ntv = ci*no*nJ;
  for(int idx=lane; idx<ntv; idx+=32){
    int c = idx/(no*nJ); int rem = idx - c*no*nJ; int o = rem/nJ; int j = rem - o*nJ;
    float sacc=0.f;
    for(int i=0;i<ni;i++) sacc += bS[wv][(o*ni+i)*nJ+j]*hsS[wv][c*ni+i];
    tS[wv][idx]=sacc;
  }
  __syncthreads();
  int nO = co*no;
  const float* Re = R + (size_t)e*Nout;
  float* oe = out + (size_t)e*nO;
  for(int idx=lane; idx<nO; idx+=32){
    int d = idx/no, o = idx - d*no;
    float sacc = 0.f;
    for(int j=0;j<nJ;j++){
      const float* Rj = Re + (j*co + d)*ci;
      for(int c=0;c<ci;c++) sacc += tS[wv][(c*no+o)*nJ+j]*Rj[c];
    }
    if(acc) sacc += oe[idx];
    oe[idx] = sacc;
  }
}

// ------------------------- attention helpers -------------------------
__device__ __forceinline__ float atomicMaxF(float* addr, float value){
  if(value >= 0.f) return __int_as_float(atomicMax((int*)addr, __float_as_int(value)));
  else             return __uint_as_float(atomicMin((unsigned int*)addr, __float_as_uint(value)));
}

__global__ void k_q(const float* __restrict__ h, const float* __restrict__ wq,
                    float* __restrict__ q, int cin, int nd, int total){
  int idx = blockIdx.x*blockDim.x+threadIdx.x;
  if(idx>=total) return;
  int node = idx/(C_MIDC*nd); int rem = idx - node*C_MIDC*nd;
  int c = rem/nd, mm = rem - c*nd;
  const float* hn = h + (size_t)node*cin*nd;
  float acc=0.f;
  for(int cc=0;cc<cin;cc++) acc += hn[cc*nd+mm]*wq[c*cin+cc];
  q[idx]=acc;
}

__global__ void k_logits(const float* __restrict__ qb, const float* __restrict__ kb,
                         const int* __restrict__ dst, int degmask, float scale,
                         float* __restrict__ logit){
  int e = blockIdx.x*blockDim.x+threadIdx.x;
  if(e>=N_EDGESC) return;
  int n = dst[e];
  float l0=0.f,l1=0.f;
  const int cumM[5]={0,1,4,9,16};
  for(int d=0;d<4;d++){
    if(!((degmask>>d)&1)) continue;
    int nd=2*d+1;
    const float* ke = kb + (size_t)N_EDGESC*C_MIDC*cumM[d] + (size_t)e*C_MIDC*nd;
    const float* qe = qb + (size_t)N_NODESC*C_MIDC*cumM[d] + (size_t)n*C_MIDC*nd;
    for(int c=0;c<C_MIDC;c++){
      float acc=0.f;
      for(int mm=0;mm<nd;mm++) acc += ke[c*nd+mm]*qe[c*nd+mm];
      if(c<4) l0+=acc; else l1+=acc;
    }
  }
  logit[e*2+0]=l0*scale; logit[e*2+1]=l1*scale;
}

__global__ void k_segmax(const float* __restrict__ lg, const int* __restrict__ dst,
                         float* __restrict__ mx){
  int idx = blockIdx.x*blockDim.x+threadIdx.x;
  if(idx>=N_EDGESC*2) return;
  int e=idx>>1, h=idx&1;
  atomicMaxF(&mx[dst[e]*2+h], lg[idx]);
}
__global__ void k_fixnf(float* __restrict__ p, int n){
  int idx = blockIdx.x*blockDim.x+threadIdx.x;
  if(idx>=n) return;
  float v=p[idx];
  if(!(v>-INFINITY && v<INFINITY)) p[idx]=0.f;
}
__global__ void k_expsum(const float* __restrict__ lg, const float* __restrict__ mx,
                         const int* __restrict__ dst, float* __restrict__ ex,
                         float* __restrict__ den){
  int idx = blockIdx.x*blockDim.x+threadIdx.x;
  if(idx>=N_EDGESC*2) return;
  int e=idx>>1, h=idx&1;
  float v = expf(lg[idx]-mx[dst[e]*2+h]);
  ex[idx]=v;
  atomicAdd(&den[dst[e]*2+h],v);
}
__global__ void k_att(const float* __restrict__ ex, const float* __restrict__ den,
                      const int* __restrict__ dst, float* __restrict__ att){
  int idx = blockIdx.x*blockDim.x+threadIdx.x;
  if(idx>=N_EDGESC*2) return;
  int e=idx>>1, h=idx&1;
  att[idx] = ex[idx]/(den[dst[e]*2+h]+EPSILONC);
}
__global__ void k_aggv(const float* __restrict__ att, const float* __restrict__ vb,
                       const int* __restrict__ dst, float* __restrict__ agg){
  int idx = blockIdx.x*blockDim.x+threadIdx.x;
  if(idx >= N_EDGESC*128) return;
  int e = idx>>7; int r = idx&127;
  const int cum[5]={0,8,32,72,128};
  int d=0; while(r >= cum[d+1]) d++;
  int nd = 2*d+1;
  int loc = r - cum[d];
  int c = loc/nd;
  float v = vb[(size_t)N_EDGESC*cum[d] + (size_t)e*C_MIDC*nd + loc];
  float a = att[e*2 + (c>>2)];
  atomicAdd(&agg[(size_t)N_NODESC*cum[d] + (size_t)dst[e]*C_MIDC*nd + loc], a*v);
}

// ------------------------- projection + norm -------------------------
__global__ void k_proj(const float* __restrict__ agg_d, const float* __restrict__ hin_d,
                       const float* __restrict__ wp, float* __restrict__ hout_d,
                       int cin, int nd, int withH, int total){
  int idx = blockIdx.x*blockDim.x+threadIdx.x;
  if(idx>=total) return;
  int node = idx/(NUM_CHC*nd); int rem = idx - node*NUM_CHC*nd;
  int ch = rem/nd, mm = rem - ch*nd;
  int ccat = C_MIDC + (withH?cin:0);
  const float* w = wp + ch*ccat;
  const float* an = agg_d + (size_t)node*C_MIDC*nd;
  float acc=0.f;
  for(int c=0;c<C_MIDC;c++) acc += w[c]*an[c*nd+mm];
  if(withH){
    const float* hn = hin_d + (size_t)node*cin*nd;
    for(int c=0;c<cin;c++) acc += w[C_MIDC+c]*hn[c*nd+mm];
  }
  hout_d[(size_t)node*NUM_CHC*nd + ch*nd + mm] = acc;
}
__global__ void k_norm(float* __restrict__ h_d, const float* __restrict__ ng,
                       const float* __restrict__ nb, int nd){
  int node = blockIdx.x*blockDim.x+threadIdx.x;
  if(node>=N_NODESC) return;
  float* hn = h_d + (size_t)node*NUM_CHC*nd;
  float nrm[NUM_CHC];
  float mu=0.f;
  for(int c=0;c<NUM_CHC;c++){
    float s=0.f; for(int mm=0;mm<nd;mm++){ float v=hn[c*nd+mm]; s+=v*v; }
    nrm[c]=sqrtf(s+EPSILONC); mu+=nrm[c];
  }
  mu *= (1.f/NUM_CHC);
  float var=0.f;
  for(int c=0;c<NUM_CHC;c++){ float dlt=nrm[c]-mu; var+=dlt*dlt; }
  var *= (1.f/NUM_CHC);
  float inv = rsqrtf(var+EPSILONC);
  for(int c=0;c<NUM_CHC;c++){
    float scale = fmaxf((nrm[c]-mu)*inv*ng[c]+nb[c], 0.f);
    float f = scale/nrm[c];
    for(int mm=0;mm<nd;mm++) hn[c*nd+mm] *= f;
  }
}

// ------------------------- output stage -------------------------
__global__ void k_deg(const int* __restrict__ dst, float* __restrict__ deg){
  int e = blockIdx.x*blockDim.x+threadIdx.x;
  if(e>=N_EDGESC) return;
  atomicAdd(&deg[dst[e]], 1.f);
}
__global__ void k_aggN(const float* __restrict__ mout, const int* __restrict__ dst,
                       float* __restrict__ aggN){
  int idx = blockIdx.x*blockDim.x+threadIdx.x;
  if(idx>=N_EDGESC*64) return;
  int e=idx>>6, d=idx&63;
  atomicAdd(&aggN[dst[e]*64+d], mout[idx]);
}
__global__ void k_h0(const float* __restrict__ aggN, const float* __restrict__ degv,
                     const float* __restrict__ h0in, const float* __restrict__ wself,
                     float* __restrict__ h0){
  int idx = blockIdx.x*blockDim.x+threadIdx.x;
  if(idx>=N_NODESC*64) return;
  int n=idx>>6, d=idx&63;
  float acc = aggN[idx]/fmaxf(degv[n],1.f);
  const float* hn = h0in + n*16;
  for(int c=0;c<16;c++) acc += wself[d*16+c]*hn[c];
  h0[idx]=acc;
}
__global__ void k_pool(const float* __restrict__ h0, const int* __restrict__ gid,
                       float* __restrict__ pooled){
  int idx = blockIdx.x*blockDim.x+threadIdx.x;
  if(idx>=N_NODESC*64) return;
  int n=idx>>6, d=idx&63;
  atomicMaxF(&pooled[gid[n]*64+d], h0[idx]);
}
__global__ void k_fc(const float* __restrict__ pooled, const float* __restrict__ w1,
                     const float* __restrict__ b1, const float* __restrict__ w2,
                     const float* __restrict__ b2, float* __restrict__ out){
  __shared__ float zS[64];
  int g = blockIdx.x, j = threadIdx.x;
  const float* p = pooled + g*64;
  float acc = b1[j];
  for(int i=0;i<64;i++) acc += p[i]*w1[i*64+j];
  zS[j]=fmaxf(acc,0.f);
  __syncthreads();
  if(j==0){
    float o = b2[0];
    for(int i2=0;i2<64;i2++) o += zS[i2]*w2[i2];
    out[g]=o;
  }
}
__global__ void k_fill(float* __restrict__ p, float v, size_t n){
  size_t i = (size_t)blockIdx.x*blockDim.x+threadIdx.x;
  if(i<n) p[i]=v;
}

// ------------------------- host orchestration -------------------------
struct RadP{ const float *b1,*b2,*b3,*w1,*w2,*w3; };

extern "C" void kernel_launch(void* const* d_in, const int* in_sizes, int n_in,
                              void* d_out, int out_size, void* d_ws, size_t ws_size,
                              hipStream_t stream){
  (void)in_sizes; (void)out_size;
  if(n_in < 286) return;
  const float* x         = (const float*)d_in[0];
  const float* pos       = (const float*)d_in[1];
  const float* edge_feat = (const float*)d_in[2];
  int ii = 3;
  auto FP=[&](void)->const float*{ return (const float*)d_in[ii++]; };
  auto RD=[&](void)->RadP{ RadP r; r.b1=FP(); r.b2=FP(); r.b3=FP(); r.w1=FP(); r.w2=FP(); r.w3=FP(); return r; };
  // params: 'fc' < 'layers' < 'out'  (jax pytree sorted-key order)
  const float *fc_b1=FP(), *fc_b2=FP(), *fc_w1=FP(), *fc_w2=FP();
  // layer 0: keys k < nb < ng < v < wproj < wq
  RadP k0_00 = RD();
  const float* nb0[4]; for(int d=0;d<4;d++) nb0[d]=FP();
  const float* ng0[4]; for(int d=0;d<4;d++) ng0[d]=FP();
  RadP v0[4]; for(int d=0;d<4;d++) v0[d]=RD();
  const float* wp0[4]; for(int d=0;d<4;d++) wp0[d]=FP();
  const float* wq0 = FP();
  // layer 1
  RadP k1[16]; for(int p=0;p<16;p++) k1[p]=RD();   // p = di*4+do ('di,do' sorted)
  const float* nb1[4]; for(int d=0;d<4;d++) nb1[d]=FP();
  const float* ng1[4]; for(int d=0;d<4;d++) ng1[d]=FP();
  RadP v1[16]; for(int p=0;p<16;p++) v1[p]=RD();
  const float* wp1[4]; for(int d=0;d<4;d++) wp1[d]=FP();
  const float* wq1[4]; for(int d=0;d<4;d++) wq1[d]=FP();
  // out
  RadP orad[4]; for(int d=0;d<4;d++) orad[d]=RD();
  const float* wself = FP();
  const int* src = (const int*)d_in[ii++];
  const int* dstp= (const int*)d_in[ii++];
  const int* gid = (const int*)d_in[ii++];

  // workspace layout
  float* W = (float*)d_ws;
  size_t off=0;
  auto A=[&](size_t n)->float*{ float* p=W+off; off+=(n+63)&~(size_t)63; return p; };
  float* qtab = A(QTOTAL);
  float* feat = A((size_t)N_EDGESC*5);
  float* basis= A((size_t)N_EDGESC*BSTRIDE);
  float* Rbuf = A((size_t)N_EDGESC*1024);
  float* hA   = A((size_t)N_NODESC*256);
  float* hB   = A((size_t)N_NODESC*256);
  float* qb   = A((size_t)N_NODESC*128);
  float* kb   = A((size_t)N_EDGESC*128);
  float* vb   = A((size_t)N_EDGESC*128);
  float* logit= A((size_t)N_EDGESC*2);
  float* exb  = A((size_t)N_EDGESC*2);
  float* attb = A((size_t)N_EDGESC*2);
  float* mx   = A((size_t)N_NODESC*2);
  float* den  = A((size_t)N_NODESC*2);
  float* aggb = A((size_t)N_NODESC*128);
  float* mout = A((size_t)N_EDGESC*64);
  float* aggN = A((size_t)N_NODESC*64);
  float* degb = A((size_t)N_NODESC);
  float* h0b  = A((size_t)N_NODESC*64);
  float* pooled=A(64*64);
  if(off*sizeof(float) > ws_size) return;  // insufficient workspace: bail deterministically

  const int cumM[5]={0,1,4,9,16};
  auto q_d  =[&](float* b,int d){ return b + (size_t)N_NODESC*C_MIDC*cumM[d]; };
  auto e_d  =[&](float* b,int d){ return b + (size_t)N_EDGESC*C_MIDC*cumM[d]; };
  auto h16_d=[&](float* b,int d){ return b + (size_t)N_NODESC*NUM_CHC*cumM[d]; };
  auto fill=[&](float* p, float v, size_t n){
    k_fill<<<(unsigned)((n+255)/256),256,0,stream>>>(p,v,n);
  };
  auto radial=[&](const RadP& r, int Nout){
    k_radial<<<N_EDGESC/128,256,0,stream>>>(feat,r.w1,r.b1,r.w2,r.b2,r.w3,r.b3,Nout,Rbuf);
  };
  auto softmax_agg=[&](int degmask, float scale){
    k_logits<<<(N_EDGESC+255)/256,256,0,stream>>>(qb,kb,dstp,degmask,scale,logit);
    fill(mx,-INFINITY,(size_t)N_NODESC*2);
    k_segmax<<<(N_EDGESC*2+255)/256,256,0,stream>>>(logit,dstp,mx);
    k_fixnf<<<(N_NODESC*2+255)/256,256,0,stream>>>(mx,N_NODESC*2);
    fill(den,0.f,(size_t)N_NODESC*2);
    k_expsum<<<(N_EDGESC*2+255)/256,256,0,stream>>>(logit,mx,dstp,exb,den);
    k_att<<<(N_EDGESC*2+255)/256,256,0,stream>>>(exb,den,dstp,attb);
    fill(aggb,0.f,(size_t)N_NODESC*128);
    k_aggv<<<(N_EDGESC*128)/256,256,0,stream>>>(attb,vb,dstp,aggb);
  };

  // precompute Q table + basis/feat
  k_init_q<<<1,64,0,stream>>>(qtab);
  k_basis<<<N_EDGESC/64,64,0,stream>>>(pos,src,dstp,edge_feat,qtab,basis,feat);

  // ======== layer 0 (in_degs={0}, c_in=32) ========
  { int nd=1, total=N_NODESC*C_MIDC*nd;
    k_q<<<(total+255)/256,256,0,stream>>>(x,wq0,q_d(qb,0),32,nd,total); }
  radial(k0_00, 256);
  k_pairmsg<<<N_EDGESC/PM_WAVES,256,0,stream>>>(basis,b_off(0,0),x,src,Rbuf,256,e_d(kb,0),0,0,0,32,8);
  for(int dd=0;dd<4;dd++){
    radial(v0[dd], 256);
    k_pairmsg<<<N_EDGESC/PM_WAVES,256,0,stream>>>(basis,b_off(0,dd),x,src,Rbuf,256,e_d(vb,dd),0,0,dd,32,8);
  }
  softmax_agg(0x1, 0.5f);                   // dk = 4 -> 1/sqrt(dk)=0.5
  for(int d=0;d<4;d++){
    int nd=2*d+1, total=N_NODESC*NUM_CHC*nd;
    k_proj<<<(total+255)/256,256,0,stream>>>(q_d(aggb,d), d==0?x:(const float*)nullptr,
                                             wp0[d], h16_d(hA,d), 32, nd, d==0?1:0, total);
    k_norm<<<(N_NODESC+255)/256,256,0,stream>>>(h16_d(hA,d), ng0[d], nb0[d], nd);
  }

  // ======== layer 1 (in_degs={0..3}, c_in=16) ========
  for(int d=0;d<4;d++){
    int nd=2*d+1, total=N_NODESC*C_MIDC*nd;
    k_q<<<(total+255)/256,256,0,stream>>>(h16_d(hA,d),wq1[d],q_d(qb,d),16,nd,total);
  }
  for(int dd=0;dd<4;dd++) for(int di=0;di<4;di++){
    int Nout = njd(di,dd)*C_MIDC*NUM_CHC;
    radial(k1[di*4+dd], Nout);
    k_pairmsg<<<N_EDGESC/PM_WAVES,256,0,stream>>>(basis,b_off(di,dd),h16_d(hA,di),src,Rbuf,Nout,
                                                  e_d(kb,dd), di==0?0:1, di,dd,16,8);
  }
  for(int dd=0;dd<4;dd++) for(int di=0;di<4;di++){
    int Nout = njd(di,dd)*C_MIDC*NUM_CHC;
    radial(v1[di*4+dd], Nout);
    k_pairmsg<<<N_EDGESC/PM_WAVES,256,0,stream>>>(basis,b_off(di,dd),h16_d(hA,di),src,Rbuf,Nout,
                                                  e_d(vb,dd), di==0?0:1, di,dd,16,8);
  }
  softmax_agg(0xF, 0.125f);                 // dk = 64 -> 1/8
  for(int d=0;d<4;d++){
    int nd=2*d+1, total=N_NODESC*NUM_CHC*nd;
    k_proj<<<(total+255)/256,256,0,stream>>>(q_d(aggb,d), h16_d(hA,d),
                                             wp1[d], h16_d(hB,d), 16, nd, 1, total);
    k_norm<<<(N_NODESC+255)/256,256,0,stream>>>(h16_d(hB,d), ng1[d], nb1[d], nd);
  }

  // ======== output stage ========
  for(int di=0;di<4;di++){
    radial(orad[di], 1024);                 // nJ=1, co=64, ci=16
    k_pairmsg<<<N_EDGESC/PM_WAVES,256,0,stream>>>(basis,b_off(di,0),h16_d(hB,di),src,Rbuf,1024,
                                                  mout, di==0?0:1, di,0,16,64);
  }
  fill(degb,0.f,(size_t)N_NODESC);
  fill(aggN,0.f,(size_t)N_NODESC*64);
  k_deg<<<(N_EDGESC+255)/256,256,0,stream>>>(dstp,degb);
  k_aggN<<<(N_EDGESC*64)/256,256,0,stream>>>(mout,dstp,aggN);
  k_h0<<<(N_NODESC*64)/256,256,0,stream>>>(aggN,degb,h16_d(hB,0),wself,h0b);
  fill(pooled,-INFINITY,64*64);
  k_pool<<<(N_NODESC*64)/256,256,0,stream>>>(h0b,gid,pooled);
  k_fixnf<<<(64*64+255)/256,256,0,stream>>>(pooled,64*64);
  k_fc<<<N_GRAPHSC,64,0,stream>>>(pooled,fc_w1,fc_b1,fc_w2,fc_b2,(float*)d_out);
}
